// Conv1d_82523501625769
// MI455X (gfx1250) — matrix operand order repaired and checked by reading, not compiled, not run on hardware
//
#include <hip/hip_runtime.h>
#include <hip/hip_bf16.h>

typedef __attribute__((ext_vector_type(16))) __bf16 v16bf;
typedef __attribute__((ext_vector_type(8)))  __bf16 v8bf;
typedef __attribute__((ext_vector_type(2)))  __bf16 v2bf;
typedef __attribute__((ext_vector_type(8)))  float  v8f;

#define B_  8
#define C_  256
#define T_  4096
#define H_  512
#define K_  4

#define APAD 40   // LDS row pad: 80 B = 20 banks -> conflict-free
#define BPAD 40

// Block tile: 128 (H) x 128 (T). 256 threads = 8 waves (wave32).
// Wave w computes rows [w*16, w*16+16) x 128 cols -> 8 WMMA accum tiles.
__global__ __launch_bounds__(256)
void conv1d_wmma_kernel(const float* __restrict__ x,
                        const float* __restrict__ W,
                        const float* __restrict__ bias,
                        float* __restrict__ out) {
    // A: W chunk split into 4 k-planes, [k][h_local][c_local], row-padded.
    __shared__ __bf16 lds_a[K_][128][APAD];      // 40 KB
    // B: x chunk TRANSPOSED to [t_local][c_local]; t_local = t - (t0-3).
    __shared__ __bf16 lds_bT[132][BPAD];         // 10.3 KB

    const int tid  = threadIdx.x;
    const int wave = tid >> 5;
    const int lane = tid & 31;

    const int t0 = blockIdx.x * 128;
    const int h0 = blockIdx.y * 128;
    const int bi = blockIdx.z;

    const int m   = lane & 15;
    const int kba = (lane < 16) ? 0 : 8;    // A fragment K base (interleaved runs)
    const int kbb = (lane < 16) ? 0 : 8;    // B fragment K base (interleaved runs)

    v8f acc[8];
#pragma unroll
    for (int i = 0; i < 8; ++i)
#pragma unroll
        for (int r = 0; r < 8; ++r) acc[i][r] = 0.0f;

    // B-fragment loader: two 16 B runs per lane (K kbb..+7, kbb+16..+23) from transposed tile.
    auto loadB = [&](int k, int nt) -> v16bf {
        const int col = nt * 16 + m + k;     // causal shift folded into row idx
        const __bf16* brow = &lds_bT[col][kbb];
        const v8bf lo = *reinterpret_cast<const v8bf*>(brow);
        const v8bf hi = *reinterpret_cast<const v8bf*>(brow + 16);
        return __builtin_shufflevector(lo, hi, 0, 1, 2, 3, 4, 5, 6, 7,
                                       8, 9, 10, 11, 12, 13, 14, 15);
    };

    for (int c0 = 0; c0 < C_; c0 += 32) {
        // ---- Stage W chunk: rows h0..h0+127, cols (c0*4 .. c0*4+127) f32.
        // Channel-pair per thread -> packed b32 LDS writes (v_cvt_pk path).
#pragma unroll
        for (int i = 0; i < 8; ++i) {
            int idx = tid + i * 256;             // 0 .. 2047
            int row = idx >> 4;                  // 0 .. 127
            int cp  = idx & 15;                  // channel pair (2*cp, 2*cp+1)
            const float* wrow = W + (size_t)(h0 + row) * (C_ * K_) + (size_t)c0 * K_;
            const float4 v0 = *reinterpret_cast<const float4*>(wrow + (2 * cp)     * 4);
            const float4 v1 = *reinterpret_cast<const float4*>(wrow + (2 * cp + 1) * 4);
#pragma unroll
            for (int k = 0; k < K_; ++k) {
                v2bf p;
                p[0] = (__bf16)(&v0.x)[k];
                p[1] = (__bf16)(&v1.x)[k];
                *reinterpret_cast<v2bf*>(&lds_a[k][row][2 * cp]) = p;
            }
        }

        // ---- Stage x chunk transposed: lds_bT[p][c], p = t - (t0-3), 131 rows.
        for (int idx = tid; idx < 16 * 132; idx += 256) {
            int cp = idx / 132;                  // channel pair index 0..15
            int p  = idx - cp * 132;             // 0 .. 131
            int t  = t0 - 3 + p;
            v2bf v; v[0] = (__bf16)0.0f; v[1] = (__bf16)0.0f;
            if (p < 131 && t >= 0) {
                const float* xb = x + (size_t)bi * C_ * T_ + (size_t)t;
                v[0] = (__bf16)xb[(size_t)(c0 + 2 * cp)     * T_];
                v[1] = (__bf16)xb[(size_t)(c0 + 2 * cp + 1) * T_];
            }
            *reinterpret_cast<v2bf*>(&lds_bT[p][2 * cp]) = v;
        }

        __syncthreads();

        // ---- Preload the 4 A fragments (one per tap); ISA layout is two
        // contiguous 16 B runs per lane: K {0-7,16-23} / {8-15,24-31}.
        v16bf afrag[K_];
#pragma unroll
        for (int k = 0; k < K_; ++k) {
            const __bf16* arow = &lds_a[k][wave * 16 + m][0];
            const v8bf lo = *reinterpret_cast<const v8bf*>(arow + kba);
            const v8bf hi = *reinterpret_cast<const v8bf*>(arow + kba + 16);
            afrag[k] = __builtin_shufflevector(lo, hi, 0, 1, 2, 3, 4, 5, 6, 7,
                                               8, 9, 10, 11, 12, 13, 14, 15);
        }

        // ---- 32 WMMAs, software-pipelined: B fragment for step i+1 is in
        // flight while WMMA i executes (wait becomes dscnt<=2, not 0).
        v16bf bcur = loadB(0, 0);
#pragma unroll
        for (int k = 0; k < K_; ++k) {
#pragma unroll
            for (int nt = 0; nt < 8; ++nt) {
                v16bf bnext = bcur;
                if (!(k == K_ - 1 && nt == 7)) {
                    const int nk  = (nt == 7) ? k + 1 : k;
                    const int nnt = (nt == 7) ? 0 : nt + 1;
                    bnext = loadB(nk, nnt);
                }
                acc[nt] = __builtin_amdgcn_wmma_f32_16x16x32_bf16(
                    false, afrag[k], false, bcur, (short)0, acc[nt], false, false);
                bcur = bnext;
            }
        }

        __syncthreads();
    }

    // ---- Epilogue: add bias, store f32.
    // C/D layout: vgpr r, lanes 0-15 -> (M=r, N=lane); lanes 16-31 -> (M=8+r).
    const int mb = (lane >> 4) * 8;
    float bv[8];
#pragma unroll
    for (int r = 0; r < 8; ++r)
        bv[r] = bias[h0 + wave * 16 + mb + r];

#pragma unroll
    for (int nt = 0; nt < 8; ++nt) {
        const int t = t0 + nt * 16 + m;
#pragma unroll
        for (int r = 0; r < 8; ++r) {
            const int h = h0 + wave * 16 + mb + r;
            out[((size_t)bi * H_ + h) * T_ + t] = acc[nt][r] + bv[r];
        }
    }
}

extern "C" void kernel_launch(void* const* d_in, const int* in_sizes, int n_in,
                              void* d_out, int out_size, void* d_ws, size_t ws_size,
                              hipStream_t stream) {
    const float* x    = (const float*)d_in[0];
    const float* W    = (const float*)d_in[1];
    const float* bias = (const float*)d_in[2];
    float* out        = (float*)d_out;

    dim3 grid(T_ / 128, H_ / 128, B_);   // (32, 4, 8)
    dim3 block(256);
    conv1d_wmma_kernel<<<grid, block, 0, stream>>>(x, W, bias, out);
}
